// RNN_46858093200011
// MI455X (gfx1250) — matrix operand order repaired and checked by reading, not compiled, not run on hardware
//
#include <hip/hip_runtime.h>
#include <hip/hip_bf16.h>
#include <stdint.h>

#define V_   50000
#define DW_  300
#define DH_  50
#define OUT_ 4
#define B_   1024
#define T_   200
#define G4   200      // 4*DH
#define GP   208      // gate width padded to 16
#define KP   320      // DW padded to 32
#define KH   64       // DH padded to 32

typedef __attribute__((ext_vector_type(16))) __bf16         v16bf;
typedef __attribute__((ext_vector_type(8)))  float          v8f;
typedef __attribute__((ext_vector_type(16))) unsigned short v16u16;
typedef __attribute__((ext_vector_type(8)))  unsigned short v8u16;
typedef int v4i_ __attribute__((vector_size(16)));           // matches builtin param type
typedef __attribute__((address_space(1))) v4i_* gptr128;
typedef __attribute__((address_space(3))) v4i_* lptr128;

#if defined(__gfx1250__) && \
    __has_builtin(__builtin_amdgcn_global_load_async_to_lds_b128) && \
    __has_builtin(__builtin_amdgcn_s_wait_asynccnt)
#define HAVE_ASYNC_LDS 1
#else
#define HAVE_ASYNC_LDS 0
#endif

__device__ __forceinline__ unsigned short f2bf(float f){
  unsigned int u = __float_as_uint(f);
  u += 0x7FFFu + ((u >> 16) & 1u);          // round-to-nearest-even
  return (unsigned short)(u >> 16);
}
__device__ __forceinline__ float bf2f(unsigned short h){
  return __uint_as_float(((unsigned int)h) << 16);
}
__device__ __forceinline__ v8f wmma_bf16(v16bf a, v16bf b, v8f c){
  return __builtin_amdgcn_wmma_f32_16x16x32_bf16(false, a, false, b, (short)0, c, false, false);
}
__device__ __forceinline__ v16bf combine16(v8u16 lo, v8u16 hi){
  v16u16 t;
  #pragma unroll
  for(int i=0;i<8;i++){ t[i]=lo[i]; t[i+8]=hi[i]; }
  return __builtin_bit_cast(v16bf, t);
}

// ---------------- prep: fp32 -> padded bf16 tables ----------------
__global__ void k_prep_emb(const float* __restrict__ emb, unsigned short* __restrict__ embB){
  int e = blockIdx.x*256 + threadIdx.x;
  if (e >= V_*KP) return;
  int v = e / KP, k = e - v*KP;
  float f = (k < DW_) ? emb[(size_t)v*DW_ + k] : 0.f;
  embB[e] = f2bf(f);
}

__global__ void k_prep_w(const float* __restrict__ w_ih, const float* __restrict__ w_hh,
                         const float* __restrict__ b_ih, const float* __restrict__ b_hh,
                         unsigned short* __restrict__ wihB, unsigned short* __restrict__ whhB,
                         float* __restrict__ biasB){
  int e = blockIdx.x*256 + threadIdx.x;
  if (e < GP*KP){
    int n = e / KP, k = e - n*KP;
    float f = (n < G4 && k < DW_) ? w_ih[n*DW_ + k] : 0.f;
    wihB[e] = f2bf(f);
  }
  int e2 = e - GP*KP;
  if (e2 >= 0 && e2 < GP*KH){
    int n = e2 / KH, k = e2 - n*KH;
    float f = (n < G4 && k < DH_) ? w_hh[n*DH_ + k] : 0.f;
    whhB[e2] = f2bf(f);
  }
  int e3 = e - GP*KP - GP*KH;
  if (e3 >= 0 && e3 < GP){
    biasB[e3] = (e3 < G4) ? (b_ih[e3] + b_hh[e3]) : 0.f;
  }
}

// ---------------- kernel 1: Gx[t][b][GP] = emb[x] @ w_ih^T + bias ----------------
// 128 rows / block (4 waves, 2 M-subtiles per wave) so each B fragment feeds 2 WMMAs.
__global__ void __launch_bounds__(128)
k_inproj(const unsigned short* __restrict__ embB, const unsigned short* __restrict__ wihB,
         const float* __restrict__ biasB, const int* __restrict__ x, float* __restrict__ gx){
  __shared__ unsigned short At[128][KP];       // 80 KB gathered A tile (bf16)
  const int tid  = threadIdx.x;
  const int lane = tid & 31;
  const int wv   = tid >> 5;
  const int m0   = blockIdx.x * 128;           // m = t*B + b

  { // gather 128 embedding rows -> LDS, 1 thread/row
    int m = m0 + tid;
    int t = m >> 10;                           // /B_ (1024)
    int b = m & 1023;
    int idx = x[b*T_ + t];
#if HAVE_ASYNC_LDS
    gptr128 gp = (gptr128)(uintptr_t)(embB + (size_t)idx*KP);
    lptr128 lp = (lptr128)(unsigned int)(uintptr_t)&At[tid][0];
    #pragma unroll
    for(int i=0;i<40;i++)
      __builtin_amdgcn_global_load_async_to_lds_b128(gp + i, lp + i, 0, 0);
    __builtin_amdgcn_s_wait_asynccnt(0);
#else
    const uint4* src = (const uint4*)(embB + (size_t)idx*KP);
    uint4* dst = (uint4*)&At[tid][0];
    #pragma unroll
    for(int i=0;i<40;i++) dst[i] = src[i];
#endif
  }
  __syncthreads();

  v8f acc[2][13];
  #pragma unroll
  for(int nt=0;nt<13;nt++){
    float bv = biasB[nt*16 + (lane&15)];       // bias broadcast along rows
    #pragma unroll
    for(int r=0;r<8;r++){ acc[0][nt][r]=bv; acc[1][nt][r]=bv; }
  }

  const int ksel  = (lane>=16)?8:0;            // A-frag per-lane K phase
  const int kbsel = (lane>=16)?8:0;            // B-frag per-lane K phase
  const int r0 = (wv*2+0)*16 + (lane&15);
  const int r1 = (wv*2+1)*16 + (lane&15);
  for(int kt=0;kt<10;kt++){
    const int k0 = kt*32;
    v16bf a0 = combine16(*(const v8u16*)&At[r0][k0+ksel], *(const v8u16*)&At[r0][k0+16+ksel]);
    v16bf a1 = combine16(*(const v8u16*)&At[r1][k0+ksel], *(const v8u16*)&At[r1][k0+16+ksel]);
    #pragma unroll
    for(int nt=0;nt<13;nt++){
      const unsigned short* bp = wihB + (size_t)(nt*16 + (lane&15))*KP + k0 + kbsel;
      v16bf bfr = combine16(*(const v8u16*)bp, *(const v8u16*)(bp + 16));
      acc[0][nt] = wmma_bf16(a0, bfr, acc[0][nt]);   // one B fragment, two WMMAs
      acc[1][nt] = wmma_bf16(a1, bfr, acc[1][nt]);
    }
  }

  #pragma unroll
  for(int ms=0; ms<2; ms++){
    const int rb = m0 + (wv*2+ms)*16 + ((lane>=16)?8:0);
    #pragma unroll
    for(int nt=0;nt<13;nt++){
      const int col = nt*16 + (lane&15);
      #pragma unroll
      for(int r=0;r<8;r++)
        gx[(size_t)(rb + r)*GP + col] = acc[ms][nt][r];
    }
  }
}

// ---------------- kernel 2: sequential LSTM over T, 16 batch rows per WG ----------------
// 4 waves share the 16-row h tile; wave w owns N-tiles {w, w+4, w+8, (w+12)}.
// w_hh fragments live in VGPRs for all 200 steps; next step's Gx slice is prefetched.
__global__ void __launch_bounds__(128)
k_recur(const float* __restrict__ gx, const unsigned short* __restrict__ whhB,
        const int* __restrict__ lengs, const float* __restrict__ w_out,
        const float* __restrict__ b_out, float* __restrict__ out){
  __shared__ unsigned short Hs[16][KH];        // h state, bf16, K-padded
  __shared__ float          Cs[16][KH];        // c state, f32
  __shared__ float          Gs[16][GP];        // gate pre-activations, f32
  __shared__ int            Ls[16];
  const int tid = threadIdx.x, lane = tid&31, wv = tid>>5;
  const int b0  = blockIdx.x * 16;

  for(int i=tid;i<16*KH;i+=128){ Hs[i>>6][i&63]=0; Cs[i>>6][i&63]=0.f; }
  if (tid<16) Ls[tid] = lengs[b0+tid];
  __syncthreads();

  const int nnt   = (wv==0)?4:3;               // wave-uniform -> EXEC stays full
  const int kbsel = (lane>=16)?8:0;
  v16bf Bf[4][2];
  #pragma unroll
  for(int q=0;q<4;q++){
    const int nt = wv + 4*q;
    if (q < nnt){
      #pragma unroll
      for(int kt=0;kt<2;kt++){
        const unsigned short* bp = whhB + (size_t)(nt*16 + (lane&15))*KH + kt*32 + kbsel;
        Bf[q][kt] = combine16(*(const v8u16*)bp, *(const v8u16*)(bp + 16));
      }
    }
  }

  const int mrow = lane&15;
  const int ksel = (lane>=16)?8:0;
  const int rb   = (lane>=16)?8:0;

  for(int t=0;t<T_;t++){
    const float* gxt = gx + (size_t)t*(B_*GP) + (size_t)b0*GP;
    v8f acc[4];
    #pragma unroll
    for(int q=0;q<4;q++) if (q<nnt){
      const int col = (wv+4*q)*16 + (lane&15);
      #pragma unroll
      for(int r=0;r<8;r++) acc[q][r] = gxt[(size_t)(rb+r)*GP + col];
    }
    // prefetch next timestep's Gx slice (overlaps WMMA + gate math + barriers)
    if (t+1 < T_){
      const char* nxt = (const char*)(gx + (size_t)(t+1)*(B_*GP) + (size_t)b0*GP);
      const int offb = tid*128;
      if (offb < 16*GP*4) __builtin_prefetch(nxt + offb, 0, 3);
    }
    #pragma unroll
    for(int kt=0;kt<2;kt++){
      const int k0 = kt*32;
      v8u16 alo = *(const v8u16*)&Hs[mrow][k0 + ksel];
      v8u16 ahi = *(const v8u16*)&Hs[mrow][k0 + 16 + ksel];
      v16bf a = combine16(alo, ahi);
      #pragma unroll
      for(int q=0;q<4;q++) if (q<nnt) acc[q] = wmma_bf16(a, Bf[q][kt], acc[q]);
    }
    #pragma unroll
    for(int q=0;q<4;q++) if (q<nnt){
      const int col = (wv+4*q)*16 + (lane&15);
      #pragma unroll
      for(int r=0;r<8;r++) Gs[rb+r][col] = acc[q][r];
    }
    __syncthreads();

    for(int e=tid;e<16*DH_;e+=128){
      const int bl = e/DH_, j = e - bl*DH_;
      const float gi = Gs[bl][j],        gf = Gs[bl][j+DH_];
      const float gg = Gs[bl][j+2*DH_],  go = Gs[bl][j+3*DH_];
      const float c_old = Cs[bl][j];
      const float si = 1.f/(1.f+__expf(-gi));
      const float sf = 1.f/(1.f+__expf(-gf));
      const float so = 1.f/(1.f+__expf(-go));
      const float cn = sf*c_old + si*tanhf(gg);
      const float hn = so*tanhf(cn);
      const bool keep = t < Ls[bl];
      const float h2 = keep ? hn : bf2f(Hs[bl][j]);
      const float c2 = keep ? cn : c_old;
      Cs[bl][j] = c2;
      Hs[bl][j] = f2bf(h2);
    }
    __syncthreads();
  }

  // fused logits + softmax (OUT=4, tiny)
  if (tid < 16){
    float lg[4];
    #pragma unroll
    for(int o=0;o<OUT_;o++){
      float s = b_out[o];
      for(int j=0;j<DH_;j++) s += bf2f(Hs[tid][j]) * w_out[o*DH_ + j];
      lg[o] = s;
    }
    float mx = fmaxf(fmaxf(lg[0],lg[1]), fmaxf(lg[2],lg[3]));
    float e0=__expf(lg[0]-mx), e1=__expf(lg[1]-mx), e2=__expf(lg[2]-mx), e3=__expf(lg[3]-mx);
    float s = e0+e1+e2+e3;
    float* po = out + (size_t)(b0+tid)*OUT_;
    po[0]=e0/s; po[1]=e1/s; po[2]=e2/s; po[3]=e3/s;
  }
}

// ---------------- launcher ----------------
extern "C" void kernel_launch(void* const* d_in, const int* in_sizes, int n_in,
                              void* d_out, int out_size, void* d_ws, size_t ws_size,
                              hipStream_t stream){
  (void)in_sizes; (void)n_in; (void)out_size;
  const int*   x     = (const int*)  d_in[0];
  const int*   lengs = (const int*)  d_in[1];
  const float* emb   = (const float*)d_in[2];
  const float* w_ih  = (const float*)d_in[3];
  const float* w_hh  = (const float*)d_in[4];
  const float* b_ih  = (const float*)d_in[5];
  const float* b_hh  = (const float*)d_in[6];
  const float* w_out = (const float*)d_in[7];
  const float* b_out = (const float*)d_in[8];
  float* out = (float*)d_out;

  uint8_t* ws = (uint8_t*)d_ws;
  size_t off = 0;
  auto alloc = [&](size_t bytes)->uint8_t*{
    uint8_t* p = ws + off;
    off += (bytes + 255) & ~(size_t)255;
    return p;
  };
  unsigned short* embB  = (unsigned short*)alloc((size_t)V_*KP*2);    // 32.0 MB
  unsigned short* wihB  = (unsigned short*)alloc((size_t)GP*KP*2);    // 133 KB
  unsigned short* whhB  = (unsigned short*)alloc((size_t)GP*KH*2);    // 27 KB
  float*          biasB = (float*)         alloc((size_t)GP*4);
  float*          gxbuf = (float*)         alloc((size_t)T_*B_*GP*4); // 170.4 MB
  if (off > ws_size) return;   // workspace too small: do nothing (deterministic)

  k_prep_emb<<<(V_*KP + 255)/256, 256, 0, stream>>>(emb, embB);
  const int prepN = GP*KP + GP*KH + GP;
  k_prep_w<<<(prepN + 255)/256, 256, 0, stream>>>(w_ih, w_hh, b_ih, b_hh, wihB, whhB, biasB);
  k_inproj<<<(T_*B_)/128, 128, 0, stream>>>(embB, wihB, biasB, x, gxbuf);
  k_recur<<<B_/16, 128, 0, stream>>>(gxbuf, whhB, lengs, w_out, b_out, out);
}
